// WinTransformerBlock_8177617731874
// MI455X (gfx1250) — matrix operand order repaired and checked by reading, not compiled, not run on hardware
//
#include <hip/hip_runtime.h>

typedef _Float16 half_t;
typedef __attribute__((ext_vector_type(16))) _Float16 v16h;
typedef __attribute__((ext_vector_type(8)))  float    v8f;
typedef __attribute__((ext_vector_type(4)))  unsigned int u32x4;
typedef __attribute__((ext_vector_type(8)))  int          i32x8;
typedef __attribute__((ext_vector_type(4)))  int          i32x4;

#define NH_    4
#define DIM_   96
#define HD_    24
#define NTOK   64
#define ATT_LD 66

// ---------------- TDM: tensor_load_to_lds (toolchain-dependent arity) --------
#if defined(__clang_major__) && (__clang_major__ >= 23)
#define TDM_LOAD(g0, g1, g2, g3) \
  __builtin_amdgcn_tensor_load_to_lds((g0), (g1), (g2), (g3), (i32x8){0,0,0,0,0,0,0,0}, 0)
#else
#define TDM_LOAD(g0, g1, g2, g3) \
  __builtin_amdgcn_tensor_load_to_lds((g0), (g1), (g2), (g3), 0)
#endif

// ---------------- WMMA fragment loaders (CDNA5 16x16x32 f16 layouts) ---------

union FragU  { v16h h; unsigned int u[8]; };
union PairU  { unsigned int u; _Float16 h[2]; };

// A 16x32 (MxK), row-major [M][ld]: lane m = lane&15; v<4: K=2v+8*hi, v>=4: K=16+2(v-4)+8*hi
__device__ __forceinline__ v16h load_a(const half_t* __restrict__ base, int ld,
                                       int m0, int k0, int kvalid, int lane) {
  const int m  = m0 + (lane & 15);
  const int hi = lane >> 4;
  FragU r;
#pragma unroll
  for (int v = 0; v < 8; ++v) {
    const int kb = (v < 4) ? (2 * v + 8 * hi) : (16 + 2 * (v - 4) + 8 * hi);
    const int k  = k0 + kb;
    r.u[v] = (k < kvalid)
                 ? *reinterpret_cast<const unsigned int*>(base + m * ld + k)
                 : 0u;
  }
  return r.h;
}

// B 32x16 (KxN) from K-contiguous storage: element(k,n) = base[n*ld + k]
__device__ __forceinline__ v16h load_kc(const half_t* __restrict__ base, int ld,
                                        int k0, int n0, int kvalid, int lane) {
  const int n  = n0 + (lane & 15);
  const int hi = lane >> 4;
  FragU r;
#pragma unroll
  for (int v = 0; v < 8; ++v) {
    const int k = k0 + ((v < 4) ? (2 * v + 8 * hi) : (16 + 2 * (v - 4) + 8 * hi));
    r.u[v] = (k < kvalid)
                 ? *reinterpret_cast<const unsigned int*>(base + n * ld + k)
                 : 0u;
  }
  return r.h;
}

#define WMMA(A, B, C) \
  __builtin_amdgcn_wmma_f32_16x16x32_f16(false, (A), false, (B), (short)0, (C), false, false)

#define RFL(v) __builtin_amdgcn_readfirstlane(v)

// ---------------- Kernel 0: weight f32 -> f16, transposed to [N][K] ----------

__global__ void k_prep(const float* __restrict__ qkvw, const float* __restrict__ projw,
                       const float* __restrict__ fc1w, const float* __restrict__ fc2w,
                       half_t* __restrict__ wqt, half_t* __restrict__ wpt,
                       half_t* __restrict__ w1t, half_t* __restrict__ w2t) {
  const int i = blockIdx.x * 256 + threadIdx.x;
  if (i < 288 * 96) {
    const int n = i / 96, k = i % 96;
    wqt[i] = (half_t)qkvw[k * 288 + n];
  }
  if (i < 96 * 96) {
    const int n = i / 96, k = i % 96;
    wpt[i] = (half_t)projw[k * 96 + n];
  }
  if (i < 384 * 96) {
    const int n = i / 96, k = i % 96;
    w1t[i] = (half_t)fc1w[k * 384 + n];
  }
  if (i < 96 * 384) {
    const int n = i / 384, k = i % 384;
    w2t[i] = (half_t)fc2w[k * 96 + n];
  }
}

// ---------------- Kernel 1: fused window attention ---------------------------
// smem carve (bytes):
//   xs  f32 [96][64] @ 0      (24576)  channel-major window input (TDM order)
//   att f32 [64][66] @ 24576  (16896)
//   an  f16 [64][96] @ 41472  (12288)  LN1 out / softmax probs (reuse)
//   qs  f16 [64][96] @ 53760
//   ks  f16 [64][96] @ 66048
//   vt  f16 [96][64] @ 78336           V transposed (K-contiguous for PV)
//   ob  f16 [64][96] @ 90624  (12288)
#define K1_SMEM 102912

__global__ __launch_bounds__(256) void k_attn(
    const float* __restrict__ x,
    const float* __restrict__ ln1w, const float* __restrict__ ln1b,
    const float* __restrict__ qkvb, const float* __restrict__ rbt,
    const float* __restrict__ projb,
    const half_t* __restrict__ wqt, const half_t* __restrict__ wpt,
    float* __restrict__ y) {
  extern __shared__ char smem[];
  float*  xs  = (float*)(smem);            // [c][tok]
  float*  att = (float*)(smem + 24576);
  half_t* an  = (half_t*)(smem + 41472);
  half_t* qs  = (half_t*)(smem + 53760);
  half_t* ks  = (half_t*)(smem + 66048);
  half_t* vt  = (half_t*)(smem + 78336);   // [c][tok]
  half_t* ob  = (half_t*)(smem + 90624);
  half_t* pb  = an;

  const int tid  = threadIdx.x;
  const int lane = tid & 31;
  const int wave = tid >> 5;
  const int hi   = lane >> 4;
  const int nl   = lane & 15;

  const int b  = blockIdx.x >> 10;
  const int w  = blockIdx.x & 1023;
  const int wr = w >> 5, wc = w & 31;

  // ---- Phase A: TDM 3D tile gather: 8(x) x 8(y, stride 256) x 96(c, stride 65536)
  if (tid < 32) {
    const unsigned long long ga =
        (unsigned long long)(uintptr_t)(x + (((size_t)b * DIM_) << 16) +
                                        ((size_t)(wr * 8) << 8) + (size_t)(wc * 8));
    const unsigned lds0 = (unsigned)(uintptr_t)xs;
    u32x4 g0;
    g0[0] = 1u;                                      // count=1 (valid user D#)
    g0[1] = lds0;                                    // lds_addr
    g0[2] = (unsigned)ga;                            // global_addr lo
    g0[3] = ((unsigned)(ga >> 32) & 0x1FFFFFFu) | (2u << 30);  // hi | type=2
    i32x8 g1;
    g1[0] = (int)(2u << 16);                         // data_size=4B
    g1[1] = 0;                                       // tensor_dim0 lo16
    g1[2] = (int)0x10;                               // dim0 hi16 ; dim1 lo16
    g1[3] = (int)((0x10u) | (8u << 16));             // dim1 hi16 ; tile_dim0=8
    g1[4] = (int)((8u) | (96u << 16));               // tile_dim1=8 ; tile_dim2=96
    g1[5] = 256;                                     // tensor_dim0_stride
    g1[6] = 0;
    g1[7] = 1;                                       // tensor_dim1_stride = 65536
    i32x4 g2; g2[0] = 0x100000; g2[1] = 0; g2[2] = 0; g2[3] = 0;
    i32x4 g3 = {0, 0, 0, 0};
    TDM_LOAD(g0, g1, g2, g3);
    __builtin_amdgcn_s_wait_tensorcnt(0);
  }
  __syncthreads();

  // ---- Phase B: LayerNorm 1 -> f16 (channel-major xs: conflict-free) --------
  if (tid < NTOK) {
    float mu = 0.f;
#pragma unroll 4
    for (int c = 0; c < DIM_; ++c) mu += xs[c * 64 + tid];
    mu *= (1.0f / DIM_);
    float var = 0.f;
#pragma unroll 4
    for (int c = 0; c < DIM_; ++c) {
      const float d = xs[c * 64 + tid] - mu;
      var += d * d;
    }
    const float rs = rsqrtf(var * (1.0f / DIM_) + 1e-5f);
    for (int c = 0; c < DIM_; c += 2) {
      PairU p;
      p.h[0] = (half_t)((xs[c * 64 + tid] - mu) * rs * ln1w[c] + ln1b[c]);
      p.h[1] = (half_t)((xs[(c + 1) * 64 + tid] - mu) * rs * ln1w[c + 1] + ln1b[c + 1]);
      *reinterpret_cast<unsigned int*>(an + tid * DIM_ + c) = p.u;
    }
  }
  __syncthreads();

  // ---- Phase C: QKV GEMM [64x96]@[96x288] -----------------------------------
  const float scale = rsqrtf((float)HD_);
  for (int t = wave; t < 72; t += 8) {
    const int tu = RFL(t);                 // scalarize tile index
    const int mi = tu / 18, sn = tu % 18;
    const int s  = sn / 6;                 // 0=q 1=k 2=v (scalar)
    const int m0 = mi * 16, r0 = (sn % 6) * 16;
    v8f acc = {};
#pragma unroll
    for (int kc = 0; kc < 3; ++kc) {
      v16h a  = load_a(an, DIM_, m0, kc * 32, DIM_, lane);
      v16h bb = load_kc(wqt, DIM_, kc * 32, s * DIM_ + r0, DIM_, lane);
      acc = WMMA(a, bb, acc);
    }
    const int   r    = r0 + nl;
    const float bias = qkvb[s * DIM_ + r];
    if (s == 0) {
#pragma unroll
      for (int v = 0; v < 8; ++v)
        qs[(m0 + v + 8 * hi) * DIM_ + r] = (half_t)((acc[v] + bias) * scale);
    } else if (s == 1) {
#pragma unroll
      for (int v = 0; v < 8; ++v)
        ks[(m0 + v + 8 * hi) * DIM_ + r] = (half_t)(acc[v] + bias);
    } else {
#pragma unroll
      for (int v = 0; v < 8; v += 2) {     // consecutive m -> packed b32 store
        const int m = m0 + v + 8 * hi;
        PairU p;
        p.h[0] = (half_t)(acc[v] + bias);
        p.h[1] = (half_t)(acc[v + 1] + bias);
        *reinterpret_cast<unsigned int*>(vt + r * 64 + m) = p.u;
      }
    }
  }
  __syncthreads();

  // ---- Phase D: per-head attention ------------------------------------------
  for (int h = 0; h < NH_; ++h) {
    // D1: logits = q @ k^T + rel-pos bias  (K=24 zero-padded to 32)
    for (int t = wave; t < 16; t += 8) {
      const int tu = RFL(t);
      const int m0 = (tu >> 2) * 16, n0 = (tu & 3) * 16;
      v16h a  = load_a(qs + h * HD_, DIM_, m0, 0, HD_, lane);
      v16h bb = load_kc(ks + h * HD_, DIM_, 0, n0, HD_, lane);
      v8f acc = {};
      acc = WMMA(a, bb, acc);
#pragma unroll
      for (int v = 0; v < 8; ++v) {
        const int row = m0 + v + 8 * hi;
        const int col = n0 + nl;
        const int i1 = row >> 3, j1 = row & 7, i2 = col >> 3, j2 = col & 7;
        const int ridx = (i1 - i2 + 7) * 15 + (j1 - j2 + 7);
        att[row * ATT_LD + col] = acc[v] + rbt[ridx * NH_ + h];
      }
    }
    __syncthreads();
    // D2: row softmax -> f16 probs (packed stores)
    if (tid < NTOK) {
      float mx = -1e30f;
      for (int c = 0; c < NTOK; ++c) mx = fmaxf(mx, att[tid * ATT_LD + c]);
      float sm = 0.f;
      for (int c = 0; c < NTOK; ++c) {
        const float e = __expf(att[tid * ATT_LD + c] - mx);
        att[tid * ATT_LD + c] = e;
        sm += e;
      }
      const float inv = 1.0f / sm;
      for (int c = 0; c < NTOK; c += 2) {
        PairU p;
        p.h[0] = (half_t)(att[tid * ATT_LD + c] * inv);
        p.h[1] = (half_t)(att[tid * ATT_LD + c + 1] * inv);
        *reinterpret_cast<unsigned int*>(pb + tid * NTOK + c) = p.u;
      }
    }
    __syncthreads();
    // D3: out_h = p @ v_h  (K=64, B from transposed V)
    for (int t = wave; t < 8; t += 8) {
      const int tu = RFL(t);
      const int m0 = (tu >> 1) * 16, n0 = (tu & 1) * 16;
      v8f acc = {};
#pragma unroll
      for (int kc = 0; kc < 2; ++kc) {
        v16h a  = load_a(pb, NTOK, m0, kc * 32, NTOK, lane);
        v16h bb = load_kc(vt + h * HD_ * 64, 64, kc * 32, n0, NTOK, lane);
        acc = WMMA(a, bb, acc);
      }
#pragma unroll
      for (int v = 0; v < 8; ++v) {
        const int m  = m0 + v + 8 * hi;
        const int dd = n0 + nl;
        if (dd < HD_) ob[m * DIM_ + h * HD_ + dd] = (half_t)acc[v];
      }
    }
    __syncthreads();
  }

  // ---- Phase E: proj + residual -> y [B,L,C] f32 ----------------------------
  for (int t = wave; t < 24; t += 8) {
    const int tu = RFL(t);
    const int m0 = (tu / 6) * 16, n0 = (tu % 6) * 16;
    v8f acc = {};
#pragma unroll
    for (int kc = 0; kc < 3; ++kc) {
      v16h a  = load_a(ob, DIM_, m0, kc * 32, DIM_, lane);
      v16h bb = load_kc(wpt, DIM_, kc * 32, n0, DIM_, lane);
      acc = WMMA(a, bb, acc);
    }
    const int   c  = n0 + nl;
    const float pc = projb[c];
#pragma unroll
    for (int v = 0; v < 8; ++v) {
      const int m = m0 + v + 8 * hi;
      const int i = m >> 3, j = m & 7;
      const int pix = ((wr * 8 + i) << 8) + wc * 8 + j;
      y[((b << 16) + pix) * DIM_ + c] = acc[v] + pc + xs[c * 64 + m];
    }
  }
}

// ---------------- Kernel 2: fused MLP ----------------------------------------
// smem: ys f32 [64][96] @0 (24576) | an2 f16 [64][96] @24576 | hb f16 [64][384] @36864
#define K2_SMEM 86016

__global__ __launch_bounds__(256) void k_mlp(
    const float* __restrict__ y,
    const float* __restrict__ ln2w, const float* __restrict__ ln2b,
    const float* __restrict__ fc1b, const float* __restrict__ fc2b,
    const half_t* __restrict__ w1t, const half_t* __restrict__ w2t,
    float* __restrict__ out) {
  extern __shared__ char smem[];
  float*  ys  = (float*)(smem);
  half_t* an2 = (half_t*)(smem + 24576);
  half_t* hb  = (half_t*)(smem + 36864);

  const int tid  = threadIdx.x;
  const int lane = tid & 31;
  const int wave = tid >> 5;
  const int hi   = lane >> 4;
  const int nl   = lane & 15;

  // ---- TDM load: one contiguous row of 6144 dwords --------------------------
  if (tid < 32) {
    const unsigned long long ga =
        (unsigned long long)(uintptr_t)(y + (size_t)blockIdx.x * (NTOK * DIM_));
    const unsigned lds0 = (unsigned)(uintptr_t)ys;
    u32x4 g0;
    g0[0] = 1u;
    g0[1] = lds0;
    g0[2] = (unsigned)ga;
    g0[3] = ((unsigned)(ga >> 32) & 0x1FFFFFFu) | (2u << 30);
    i32x8 g1;
    g1[0] = (int)(2u << 16);                 // data_size=4B
    g1[1] = 0;
    g1[2] = (int)0x10;
    g1[3] = (int)((0x10u) | (6144u << 16));  // tile_dim0 = 6144
    g1[4] = 1;                               // tile_dim1 = 1
    g1[5] = 6144;
    g1[6] = 0;
    g1[7] = 0;
    i32x4 gz = {0, 0, 0, 0};
    TDM_LOAD(g0, g1, gz, gz);
    __builtin_amdgcn_s_wait_tensorcnt(0);
  }
  __syncthreads();

  // ---- LN2 -> f16 ------------------------------------------------------------
  if (tid < NTOK) {
    float mu = 0.f;
#pragma unroll 4
    for (int c = 0; c < DIM_; ++c) mu += ys[tid * DIM_ + c];
    mu *= (1.0f / DIM_);
    float var = 0.f;
#pragma unroll 4
    for (int c = 0; c < DIM_; ++c) {
      const float d = ys[tid * DIM_ + c] - mu;
      var += d * d;
    }
    const float rs = rsqrtf(var * (1.0f / DIM_) + 1e-5f);
    for (int c = 0; c < DIM_; c += 2) {
      PairU p;
      p.h[0] = (half_t)((ys[tid * DIM_ + c] - mu) * rs * ln2w[c] + ln2b[c]);
      p.h[1] = (half_t)((ys[tid * DIM_ + c + 1] - mu) * rs * ln2w[c + 1] + ln2b[c + 1]);
      *reinterpret_cast<unsigned int*>(an2 + tid * DIM_ + c) = p.u;
    }
  }
  __syncthreads();

  // ---- fc1 + exact GELU: [64x96]@[96x384] -----------------------------------
  for (int t = wave; t < 96; t += 8) {
    const int tu = RFL(t);
    const int m0 = (tu / 24) * 16, n0 = (tu % 24) * 16;
    v8f acc = {};
#pragma unroll
    for (int kc = 0; kc < 3; ++kc) {
      v16h a  = load_a(an2, DIM_, m0, kc * 32, DIM_, lane);
      v16h bb = load_kc(w1t, DIM_, kc * 32, n0, DIM_, lane);
      acc = WMMA(a, bb, acc);
    }
    const int   n  = n0 + nl;
    const float fb = fc1b[n];
#pragma unroll
    for (int v = 0; v < 8; ++v) {
      const int m = m0 + v + 8 * hi;
      const float z = acc[v] + fb;
      const float g = 0.5f * z * (1.0f + erff(z * 0.70710678118654752f));
      hb[m * 384 + n] = (half_t)g;
    }
  }
  __syncthreads();

  // ---- fc2 + residual, NCHW store: [64x384]@[384x96] ------------------------
  for (int t = wave; t < 24; t += 8) {
    const int tu = RFL(t);
    const int m0 = (tu / 6) * 16, n0 = (tu % 6) * 16;
    v8f acc = {};
#pragma unroll
    for (int kc = 0; kc < 12; ++kc) {
      v16h a  = load_a(hb, 384, m0, kc * 32, 384, lane);
      v16h bb = load_kc(w2t, 384, kc * 32, n0, 384, lane);
      acc = WMMA(a, bb, acc);
    }
    const int   c  = n0 + nl;
    const float fb = fc2b[c];
#pragma unroll
    for (int v = 0; v < 8; ++v) {
      const int m = m0 + v + 8 * hi;
      const float val = acc[v] + fb + ys[m * DIM_ + c];
      const int tg  = blockIdx.x * NTOK + m;
      const int bb_ = tg >> 16;
      const int pix = tg & 65535;
      out[((bb_ * DIM_ + c) << 16) + pix] = val;
    }
  }
}

// ---------------- host launcher ----------------------------------------------

extern "C" void kernel_launch(void* const* d_in, const int* in_sizes, int n_in,
                              void* d_out, int out_size, void* d_ws, size_t ws_size,
                              hipStream_t stream) {
  const float* x     = (const float*)d_in[0];
  const float* ln1w  = (const float*)d_in[1];
  const float* ln1b  = (const float*)d_in[2];
  const float* qkvw  = (const float*)d_in[3];
  const float* qkvb  = (const float*)d_in[4];
  const float* rbt   = (const float*)d_in[5];
  const float* projw = (const float*)d_in[6];
  const float* projb = (const float*)d_in[7];
  const float* ln2w  = (const float*)d_in[8];
  const float* ln2b  = (const float*)d_in[9];
  const float* fc1w  = (const float*)d_in[10];
  const float* fc1b  = (const float*)d_in[11];
  const float* fc2w  = (const float*)d_in[12];
  const float* fc2b  = (const float*)d_in[13];
  float* out = (float*)d_out;

  char* ws = (char*)d_ws;
  float*  y   = (float*)ws;                                         // 201326592 B
  half_t* wqt = (half_t*)(ws + 201326592);                          // 288x96
  half_t* wpt = (half_t*)(ws + 201326592 + 55296);                  // 96x96
  half_t* w1t = (half_t*)(ws + 201326592 + 55296 + 18432);          // 384x96
  half_t* w2t = (half_t*)(ws + 201326592 + 55296 + 18432 + 73728);  // 96x384

  hipFuncSetAttribute((const void*)k_attn,
                      hipFuncAttributeMaxDynamicSharedMemorySize, K1_SMEM);
  hipFuncSetAttribute((const void*)k_mlp,
                      hipFuncAttributeMaxDynamicSharedMemorySize, K2_SMEM);

  k_prep<<<144, 256, 0, stream>>>(qkvw, projw, fc1w, fc2w, wqt, wpt, w1t, w2t);
  k_attn<<<8192, 256, K1_SMEM, stream>>>(x, ln1w, ln1b, qkvb, rbt, projb, wqt, wpt, y);
  k_mlp<<<8192, 256, K2_SMEM, stream>>>(y, ln2w, ln2b, fc1b, fc2b, w1t, w2t, out);
}
